// ExplainableRecurrentPointProcess_16690242913094
// MI455X (gfx1250) — matrix operand order repaired and checked by reading, not compiled, not run on hardware
//
#include <hip/hip_runtime.h>

typedef __attribute__((ext_vector_type(16))) _Float16 v16h;
typedef __attribute__((ext_vector_type(8)))  float    v8f;

#define B_  512
#define T_  512
#define E_  128
#define H_  128
#define K_  128
#define H3  384
#define TS  2048   // halfs per 16(N)x128(K) swizzled B-tile image (4 kchunks * 32 lanes * 16 halfs)

// ---- workspace layout (offsets in halfs for f16 images) ----
#define WHH_OFF   0          // 24 tiles
#define W1_OFF    49152      // 8 tiles each below
#define W2_OFF    65536
#define WF1_OFF   81920
#define WF2_OFF   98304
#define WSK_OFF   114688
#define IMG_HALFS 131072
#define IMG_BYTES (IMG_HALFS * 2)
#define G2_BYTE_OFF IMG_BYTES         // then 129*384 f32

// ---- LDS layout (offsets in halfs, then byte region) ----
#define L_WHH 0
#define L_W1  49152
#define L_W2  65536
#define L_WF1 81920
#define L_WF2 98304
#define L_WSK 114688
#define L_X0  131072
#define L_X1  133120
#define L_T1  135168
#define L_R1  137216
#define L_END_H 139264
#define L_GI_BYTE (L_END_H * 2)             // 16*384 f32
#define L_SC_BYTE (L_GI_BYTE + 16*384*4)    // 16 f32 scales
#define L_TY_BYTE (L_SC_BYTE + 64)          // 16 int types
#define L_TOTAL   (L_TY_BYTE + 64)          // 303232 bytes

// ------------------------------------------------------------------
// Phase 0a: fold embedding-lookup + W_ih + b_ih into a 129x384 table.
// Row 128 == b_ih only (zero embedding at t=0).
// ------------------------------------------------------------------
__global__ void erpp_precompute_G2(const float* __restrict__ W_embed,
                                   const float* __restrict__ W_ih,
                                   const float* __restrict__ b_ih,
                                   float* __restrict__ G2) {
  int idx = blockIdx.x * 256 + threadIdx.x;
  if (idx >= 129 * H3) return;
  int ty = idx / H3;
  int j  = idx - ty * H3;
  float acc = b_ih[j];
  if (ty < K_) {
    const float* we = W_embed + ty * E_;
    const float* wi = W_ih + j * E_;
#pragma unroll 8
    for (int e = 0; e < E_; ++e) acc += we[e] * wi[e];
  }
  G2[idx] = acc;
}

// ------------------------------------------------------------------
// Phase 0b: swizzle an [N x 128] f32 weight (used as out = h @ W^T)
// into the f16 WMMA B-operand image: [ntile][kchunk][lane][16 halfs],
// lane l(0..15)->N=l holds K=0..7,16..23, lane 16..31 holds K=8..15,24..31.
// ------------------------------------------------------------------
__global__ void erpp_swizzleB(const float* __restrict__ W,
                              _Float16* __restrict__ img, int ntiles) {
  int idx = blockIdx.x * 256 + threadIdx.x;
  if (idx >= ntiles * TS) return;
  int j    = idx & 15;
  int lane = (idx >> 4) & 31;
  int kc   = (idx >> 9) & 3;
  int nt   = idx >> 11;
  int n = nt * 16 + (lane & 15);
  int k = kc * 32 + (lane >> 4) * 8 + (j & 7) + (j >> 3) * 16;
  img[idx] = (_Float16)W[n * H_ + k];
}

// ------------------------------------------------------------------
// device helpers
// ------------------------------------------------------------------
struct Areg { v16h a[4]; };   // one 16x128 A tile held in 32 VGPRs

__device__ __forceinline__ Areg erpp_loadA(const _Float16* X) {
  int lane = threadIdx.x & 31;
  Areg A;
#pragma unroll
  for (int kc = 0; kc < 4; ++kc)
    A.a[kc] = *(const v16h*)(X + (size_t)(kc * 32 + lane) * 16);
  return A;
}

__device__ __forceinline__ v8f erpp_mmA(const Areg& A, const _Float16* Bimg, v8f acc) {
  int lane = threadIdx.x & 31;
#pragma unroll
  for (int kc = 0; kc < 4; ++kc) {
    v16h b = *(const v16h*)(Bimg + (size_t)(kc * 32 + lane) * 16);
    acc = __builtin_amdgcn_wmma_f32_16x16x32_f16(false, A.a[kc], false, b, (short)0,
                                                 acc, false, false);
  }
  return acc;
}
__device__ __forceinline__ v8f erpp_mmAz(const Areg& A, const _Float16* Bimg) {
  v8f z = {};
  return erpp_mmA(A, Bimg, z);
}

// scatter this wave's 16x16 f32 D-tile (v8f per lane) into a swizzled
// f16 A-image at columns [16w, 16w+16). A layout per ISA 7.12.2.
__device__ __forceinline__ void erpp_scatterA(_Float16* X, v8f v) {
  int lane = threadIdx.x & 31;
  int w    = threadIdx.x >> 5;
  int n    = lane & 15;
  int hi   = lane >> 4;
  int k    = w * 16 + n;          // K-column 0..127 in the A image
  int kc   = k >> 5;
  int kk   = k & 31;
  int lanehi = (kk >> 3) & 1;
  int j      = (kk & 7) + ((kk >> 4) & 1) * 8;
#pragma unroll
  for (int r = 0; r < 8; ++r) {
    int m = r + 8 * hi;
    X[(size_t)(kc * 32 + (m + 16 * lanehi)) * 16 + j] = (_Float16)v[r];
  }
}

__device__ __forceinline__ float erpp_sigmoid(float x) {
  return 1.0f / (1.0f + __expf(-x));
}

// one RK4 stage: kh = (tanh(h@W1^T+b1)@W2^T + b2)*s ; kc = exp(shallow(h))*s
__device__ __forceinline__ void erpp_fa(const _Float16* Xin, _Float16* T1A, _Float16* R1A,
                                        const _Float16* sh16, float b1c, float b2c,
                                        float bf1c, float bf2c, const float* sm,
                                        v8f& kh, v8f& kc) {
  int w = threadIdx.x >> 5;
  Areg Ax = erpp_loadA(Xin);                       // shared by W1, Wf1, Wskip
  v8f t1 = erpp_mmAz(Ax, sh16 + L_W1 + (size_t)w * TS);
  v8f r1 = erpp_mmAz(Ax, sh16 + L_WF1 + (size_t)w * TS);
  v8f t1e, r1e;
#pragma unroll
  for (int r = 0; r < 8; ++r) {
    t1e[r] = tanhf(t1[r] + b1c);
    float x = r1[r] + bf1c;
    r1e[r] = x > 0.0f ? x : 0.0f;
  }
  erpp_scatterA(T1A, t1e);
  erpp_scatterA(R1A, r1e);
  __syncthreads();
  Areg At = erpp_loadA(T1A);
  Areg Ar = erpp_loadA(R1A);
  v8f kh0 = erpp_mmAz(At, sh16 + L_W2 + (size_t)w * TS);
  v8f sh  = erpp_mmAz(Ar, sh16 + L_WF2 + (size_t)w * TS);
  sh = erpp_mmA(Ax, sh16 + L_WSK + (size_t)w * TS, sh);
#pragma unroll
  for (int r = 0; r < 8; ++r) {
    kh[r] = (kh0[r] + b2c) * sm[r];
    kc[r] = __expf(sh[r] + bf2c) * sm[r];
  }
}

// ------------------------------------------------------------------
// Main sequential scan: one block = 16 batch rows, 8 waves own 16
// output columns each. Weights resident in LDS for all 512 steps.
// ------------------------------------------------------------------
__global__ __launch_bounds__(256)
void erpp_scan(const float* __restrict__ ev, const _Float16* __restrict__ imgs,
               const float* __restrict__ G2, const float* __restrict__ b_hh,
               const float* __restrict__ b1, const float* __restrict__ b2,
               const float* __restrict__ bf1, const float* __restrict__ bf2,
               float* __restrict__ out_li, float* __restrict__ out_cum) {
  extern __shared__ __align__(128) char smem_raw[];
  _Float16* sh16    = (_Float16*)smem_raw;
  float*    gi_buf  = (float*)(smem_raw + L_GI_BYTE);
  float*    s_scale = (float*)(smem_raw + L_SC_BYTE);
  int*      s_type  = (int*)(smem_raw + L_TY_BYTE);

  const int tid  = threadIdx.x;
  const int lane = tid & 31;
  const int w    = tid >> 5;
  const int bm   = blockIdx.x;
  const int hi   = lane >> 4;
  const int c    = w * 16 + (lane & 15);   // this lane's output column

  // one-time: pull swizzled f16 weight images (256 KB) into LDS via the
  // CDNA5 async global->LDS DMA path (GVS mode, ASYNCcnt-tracked).
  {
    unsigned lds_base = (unsigned)(uintptr_t)(void*)sh16;  // low 32 bits = LDS byte offset
    unsigned long long gbase = (unsigned long long)(uintptr_t)imgs;
    for (int i = tid; i < IMG_BYTES / 16; i += 256) {
      unsigned loff = lds_base + (unsigned)i * 16u;
      unsigned goff = (unsigned)i * 16u;
      asm volatile("global_load_async_to_lds_b128 %0, %1, %2"
                   :
                   : "v"(loff), "v"(goff), "s"(gbase)
                   : "memory");
    }
    asm volatile("s_wait_asynccnt 0x0" ::: "memory");
  }
  for (int i = tid; i < TS; i += 256) sh16[L_X0 + i] = (_Float16)0.0f;  // h0 = 0
  __syncthreads();

  // per-lane column biases live in registers for all 512 steps
  const float bhhr = b_hh[c], bhhz = b_hh[H_ + c], bhhn = b_hh[2 * H_ + c];
  const float b1c = b1[c], b2c = b2[c], bf1c = bf1[c], bf2c = bf2[c];

  v8f hprev = {};
  _Float16* Xa  = sh16 + L_X0;   // image of h_prev
  _Float16* Xb  = sh16 + L_X1;
  _Float16* T1A = sh16 + L_T1;
  _Float16* R1A = sh16 + L_R1;

  for (int t = 0; t < T_; ++t) {
    // stage per-row scale s and type-folded GRU input rows
    if (tid < 16) {
      int b = bm * 16 + tid;
      float ta = (t >= 1) ? ev[((size_t)b * T_ + (t - 1)) * 2] : 0.0f;
      float tb = (t >= 2) ? ev[((size_t)b * T_ + (t - 2)) * 2] : 0.0f;
      s_scale[tid] = (t == 0) ? 0.0f : (ta - tb);
      s_type[tid]  = (t == 0) ? 128 : (int)ev[((size_t)b * T_ + (t - 1)) * 2 + 1];
      if (t + 1 < T_)
        __builtin_prefetch(&ev[((size_t)b * T_ + t) * 2], 0, 1);  // next step's row
    }
    __syncthreads();
    for (int idx = tid; idx < 16 * H3; idx += 256) {
      int m = idx / H3;
      int j = idx - m * H3;
      gi_buf[idx] = G2[(size_t)s_type[m] * H3 + j];   // = x@W_ih^T + b_ih
    }
    __syncthreads();

    float sm[8];
#pragma unroll
    for (int r = 0; r < 8; ++r) sm[r] = s_scale[r + 8 * hi];

    // ---- GRU: gh = h @ W_hh^T (3 gates share one A tile) ----
    Areg Ah = erpp_loadA(Xa);
    v8f ghr = erpp_mmAz(Ah, sh16 + L_WHH + (size_t)(w)      * TS);
    v8f ghz = erpp_mmAz(Ah, sh16 + L_WHH + (size_t)(8 + w)  * TS);
    v8f ghn = erpp_mmAz(Ah, sh16 + L_WHH + (size_t)(16 + w) * TS);
    v8f hg;
#pragma unroll
    for (int r = 0; r < 8; ++r) {
      int m = r + 8 * hi;
      float gir = gi_buf[m * H3 + c];
      float giz = gi_buf[m * H3 + H_ + c];
      float gin = gi_buf[m * H3 + 2 * H_ + c];
      float rg = erpp_sigmoid(gir + ghr[r] + bhhr);
      float zg = erpp_sigmoid(giz + ghz[r] + bhhz);
      float ng = tanhf(gin + rg * (ghn[r] + bhhn));
      hg[r] = (1.0f - zg) * ng + zg * hprev[r];
    }
    erpp_scatterA(Xb, hg);
    __syncthreads();

    // ---- RK4 over [0,1] with augmented cumulative intensity ----
    v8f acc_h = {}, acc_c = {};
    _Float16* Xcur = Xb;
    _Float16* Xoth = Xa;
    const float wgt[4]   = {1.0f, 2.0f, 2.0f, 1.0f};
    const float stepc[3] = {0.5f, 0.5f, 1.0f};
#pragma unroll
    for (int e = 0; e < 4; ++e) {
      v8f kh, kc;
      erpp_fa(Xcur, T1A, R1A, sh16, b1c, b2c, bf1c, bf2c, sm, kh, kc);
      acc_h = acc_h + kh * wgt[e];
      acc_c = acc_c + kc * wgt[e];
      if (e < 3) {
        v8f he = hg + kh * stepc[e];
        erpp_scatterA(Xoth, he);
        __syncthreads();
        _Float16* tmp = Xcur; Xcur = Xoth; Xoth = tmp;
      }
    }
    v8f hnew = hg + acc_h * (1.0f / 6.0f);
    v8f cnew = acc_c * (1.0f / 6.0f);
    erpp_scatterA(Xoth, hnew);
    __syncthreads();
    _Float16* Xh = Xoth; Xoth = Xcur; Xcur = Xh;

    // ---- log intensities = shallow(h_new); h_new A tile shared by Wf1/Wskip
    Areg Ahn = erpp_loadA(Xh);
    v8f r1 = erpp_mmAz(Ahn, sh16 + L_WF1 + (size_t)w * TS);
    v8f r1e;
#pragma unroll
    for (int r = 0; r < 8; ++r) {
      float x = r1[r] + bf1c;
      r1e[r] = x > 0.0f ? x : 0.0f;
    }
    erpp_scatterA(R1A, r1e);
    __syncthreads();
    Areg Ar1 = erpp_loadA(R1A);
    v8f li = erpp_mmAz(Ar1, sh16 + L_WF2 + (size_t)w * TS);
    li = erpp_mmA(Ahn, sh16 + L_WSK + (size_t)w * TS, li);

#pragma unroll
    for (int r = 0; r < 8; ++r) {
      int m = r + 8 * hi;
      size_t b = (size_t)bm * 16 + m;
      size_t o = (b * T_ + t) * (size_t)K_ + c;
      out_li[o]  = li[r] + bf2c;
      out_cum[o] = cnew[r];
    }
    hprev = hnew;
    Xa = Xh;                                        // next step's h_prev image
    Xb = (Xh == sh16 + L_X0) ? (sh16 + L_X1) : (sh16 + L_X0);
  }
}

// ------------------------------------------------------------------
extern "C" void kernel_launch(void* const* d_in, const int* in_sizes, int n_in,
                              void* d_out, int out_size, void* d_ws, size_t ws_size,
                              hipStream_t stream) {
  (void)in_sizes; (void)n_in; (void)out_size; (void)ws_size;
  const float* ev      = (const float*)d_in[0];
  const float* W_embed = (const float*)d_in[1];
  const float* W_ih    = (const float*)d_in[2];
  const float* W_hh    = (const float*)d_in[3];
  const float* b_ih    = (const float*)d_in[4];
  const float* b_hh    = (const float*)d_in[5];
  const float* W1      = (const float*)d_in[6];
  const float* b1      = (const float*)d_in[7];
  const float* W2      = (const float*)d_in[8];
  const float* b2      = (const float*)d_in[9];
  const float* Wf1     = (const float*)d_in[10];
  const float* bf1     = (const float*)d_in[11];
  const float* Wf2     = (const float*)d_in[12];
  const float* bf2     = (const float*)d_in[13];
  const float* Wskip   = (const float*)d_in[14];

  _Float16* imgs = (_Float16*)d_ws;
  float*    G2   = (float*)((char*)d_ws + G2_BYTE_OFF);

  erpp_precompute_G2<<<(129 * H3 + 255) / 256, 256, 0, stream>>>(W_embed, W_ih, b_ih, G2);
  erpp_swizzleB<<<(24 * TS + 255) / 256, 256, 0, stream>>>(W_hh,  imgs + WHH_OFF, 24);
  erpp_swizzleB<<<(8 * TS + 255) / 256, 256, 0, stream>>>(W1,    imgs + W1_OFF, 8);
  erpp_swizzleB<<<(8 * TS + 255) / 256, 256, 0, stream>>>(W2,    imgs + W2_OFF, 8);
  erpp_swizzleB<<<(8 * TS + 255) / 256, 256, 0, stream>>>(Wf1,   imgs + WF1_OFF, 8);
  erpp_swizzleB<<<(8 * TS + 255) / 256, 256, 0, stream>>>(Wf2,   imgs + WF2_OFF, 8);
  erpp_swizzleB<<<(8 * TS + 255) / 256, 256, 0, stream>>>(Wskip, imgs + WSK_OFF, 8);

  (void)hipFuncSetAttribute((const void*)erpp_scan,
                            hipFuncAttributeMaxDynamicSharedMemorySize, L_TOTAL);

  float* out_li  = (float*)d_out;
  float* out_cum = out_li + (size_t)B_ * T_ * K_;
  erpp_scan<<<B_ / 16, 256, L_TOTAL, stream>>>(ev, imgs, G2, b_hh, b1, b2, bf1, bf2,
                                               out_li, out_cum);
}
